// step_shGLM_16097537425995
// MI455X (gfx1250) — matrix operand order repaired and checked by reading, not compiled, not run on hardware
//
#include <hip/hip_runtime.h>
#include <hip/hip_bf16.h>
#include <math.h>

// ---- problem constants (from reference) ----
static constexpr int kSUB    = 20;
static constexpr int kE_NO   = 2000;
static constexpr int kI_NO   = 500;
static constexpr int kT_DATA = 10000;
static constexpr int kT_SYN  = 200;
static constexpr int kT_HIST = 50;
static constexpr int kB_SYN  = 3;
static constexpr int kB_HIST = 3;

static constexpr int kChunksE = (kE_NO + 31) / 32;   // 63 (62 full + tail 16)
static constexpr int kChunksI = (kI_NO + 31) / 32;   // 16 (15 full + tail 20)

// ---- workspace layout (float offsets) ----
static constexpr size_t WS_IN_E   = 0;                       // 10000*20
static constexpr size_t WS_IN_I   = WS_IN_E  + 200000;       // 10000*20
static constexpr size_t WS_SYN_S  = WS_IN_I  + 200000;       // 10000*20
static constexpr size_t WS_SYN_NS = WS_SYN_S + 200000;       // 10000*20
static constexpr size_t WS_KS     = WS_SYN_NS + 200000;      // 20*2*200
static constexpr size_t WS_KNS    = WS_KS  + 8000;           // 20*2*200
static constexpr size_t WS_HKS    = WS_KNS + 8000;           // 50*20 (reversed: [tau][s])
static constexpr size_t WS_HKNS   = WS_HKS + 1000;
static constexpr size_t WS_PKS    = WS_HKNS + 1000;
static constexpr size_t WS_PKNS   = WS_PKS + 1000;
// packed-B operand layouts (f16 stored in float slots; 32B aligned offsets)
static constexpr size_t WS_BPE    = WS_PKNS + 1000;                  // 63*1024 f16
static constexpr size_t WS_BPI    = WS_BPE + (size_t)kChunksE * 512; // 16*1024 f16

typedef __attribute__((ext_vector_type(16))) _Float16 v16h;
typedef __attribute__((ext_vector_type(8)))  _Float16 v8h;
typedef __attribute__((ext_vector_type(8)))  float    v8f;
typedef __attribute__((ext_vector_type(4)))  float    v4f;

// =====================================================================
// Kernel 1: generate synaptic / history / propagation kernel tables
// =====================================================================
__global__ void genkern_kernel(
    const float* __restrict__ W_s_syn,  const float* __restrict__ W_ns_syn,
    const float* __restrict__ Tau_s_syn,const float* __restrict__ Tau_ns_syn,
    const float* __restrict__ D_s_syn,  const float* __restrict__ D_ns_syn,
    const float* __restrict__ W_s_hist, const float* __restrict__ W_ns_hist,
    const float* __restrict__ Tau_s_hist,const float* __restrict__ Tau_ns_hist,
    const float* __restrict__ W_s_prop, const float* __restrict__ W_ns_prop,
    const float* __restrict__ Tau_s_prop,const float* __restrict__ Tau_ns_prop,
    float* __restrict__ ksS, float* __restrict__ ksNS,
    float* __restrict__ hkS, float* __restrict__ hkNS,
    float* __restrict__ pkS, float* __restrict__ pkNS)
{
    const int idx = blockIdx.x * blockDim.x + threadIdx.x;

    // synaptic kernels: idx over 20*2*200 = 8000, layout [s][c][m]
    if (idx < kSUB * 2 * kT_SYN) {
        const int s = idx / (2 * kT_SYN);
        const int c = (idx / kT_SYN) & 1;
        const int m = idx % kT_SYN;
        const float t = (float)m;
        {
            const float ts = fmaxf(t - __expf(D_s_syn[s * 2 + c]), 0.f);
            float acc = 0.f;
            #pragma unroll
            for (int b = 0; b < kB_SYN; ++b) {
                const float u = ts / __expf(Tau_s_syn[b * 2 + c]);
                acc += W_s_syn[s * 6 + b * 2 + c] * u * __expf(-u);
            }
            ksS[idx] = acc;
        }
        {
            const float ts = fmaxf(t - __expf(D_ns_syn[s * 2 + c]), 0.f);
            float acc = 0.f;
            #pragma unroll
            for (int b = 0; b < kB_SYN; ++b) {
                const float u = ts / __expf(Tau_ns_syn[b * 2 + c]);
                acc += W_ns_syn[s * 6 + b * 2 + c] * u * __expf(-u);
            }
            ksNS[idx] = acc;
        }
    }

    // history / prop kernels (time-reversed: [tau][s], tau=0 oldest)
    if (idx < kT_HIST * kSUB) {
        const int tau = idx / kSUB;
        const int s   = idx % kSUB;
        const float t = (float)(kT_HIST - 1 - tau);
        float ah = 0.f, anh = 0.f, ap = 0.f, anp = 0.f;
        #pragma unroll
        for (int b = 0; b < kB_HIST; ++b) {
            float u;
            u = t / __expf(Tau_s_hist[b]);  ah  += W_s_hist [s * 3 + b] * u * __expf(-u);
            u = t / __expf(Tau_ns_hist[b]); anh += W_ns_hist[s * 3 + b] * u * __expf(-u);
            u = t / __expf(Tau_s_prop[b]);  ap  += W_s_prop [s * 3 + b] * u * __expf(-u);
            u = t / __expf(Tau_ns_prop[b]); anp += W_ns_prop[s * 3 + b] * u * __expf(-u);
        }
        hkS[idx] = ah; hkNS[idx] = anh; pkS[idx] = ap; pkNS[idx] = anp;
    }
}

// =====================================================================
// Kernel 2a: pack B = Csyn^T into per-lane WMMA operand layout.
// Bp[ch*1024 + n*32 + k] (f16), zero-padded past n=20 and past NIN.
// =====================================================================
__global__ __launch_bounds__(256) void pack_b_kernel(
    const float* __restrict__ Csyn, _Float16* __restrict__ Bp,
    int NIN, int nChunks)
{
    const int idx = blockIdx.x * blockDim.x + threadIdx.x;
    if (idx >= nChunks * 1024) return;
    const int ch  = idx >> 10;
    const int rem = idx & 1023;
    const int n   = rem >> 5;
    const int k   = rem & 31;
    const int kk  = ch * 32 + k;
    const float v = (n < kSUB && kk < NIN) ? Csyn[(size_t)n * NIN + kk] : 0.f;
    Bp[idx] = (_Float16)v;
}

// =====================================================================
// Fragment build + 2x WMMA (shared by hot loop and tail epilogue)
// =====================================================================
__device__ __forceinline__ void wmma_step(
    const _Float16* ldsA, const _Float16* __restrict__ bb,
    int row, int half, v8f& c0, v8f& c1)
{
    // A fragment: two contiguous 8xf16 runs per lane (ISA 7.12.2 layout)
    const v8h alo = *(const v8h*)&ldsA[row * 32 + half * 8];
    const v8h ahi = *(const v8h*)&ldsA[row * 32 + 16 + half * 8];
    const v16h a = __builtin_shufflevector(alo, ahi,
        0,1,2,3,4,5,6,7,8,9,10,11,12,13,14,15);

    // B fragments: two contiguous 8xf16 runs per lane from packed global
    const v8h b0lo = *(const v8h*)&bb[row * 32 + half * 8];
    const v8h b0hi = *(const v8h*)&bb[row * 32 + 16 + half * 8];
    const v16h b0 = __builtin_shufflevector(b0lo, b0hi,
        0,1,2,3,4,5,6,7,8,9,10,11,12,13,14,15);
    const v8h b1lo = *(const v8h*)&bb[(16 + row) * 32 + half * 8];
    const v8h b1hi = *(const v8h*)&bb[(16 + row) * 32 + 16 + half * 8];
    const v16h b1 = __builtin_shufflevector(b1lo, b1hi,
        0,1,2,3,4,5,6,7,8,9,10,11,12,13,14,15);

    c0 = __builtin_amdgcn_wmma_f32_16x16x32_f16(false, a, false, b0,
                                                (short)0, c0, false, false);
    c1 = __builtin_amdgcn_wmma_f32_16x16x32_f16(false, a, false, b1,
                                                (short)0, c1, false, false);
}

// =====================================================================
// Kernel 2b: WMMA projection  out(T,20) = S(T,NIN) @ Csyn(20,NIN)^T
// Full K-chunks staged unguarded+vectorized through LDS; single
// partial tail chunk handled in a guarded epilogue.
// =====================================================================
__global__ __launch_bounds__(32) void proj_wmma_kernel(
    const float* __restrict__ S,          // (kT_DATA, NIN) row-major
    const _Float16* __restrict__ Bp,      // packed B, 1024 f16 per chunk
    float* __restrict__ out,              // (kT_DATA, kSUB)
    int NIN, int nChunks)
{
    __shared__ _Float16 ldsA[16 * 32];    // [r][k] f16 tile

    const int lane = threadIdx.x;         // 0..31
    const int row  = lane & 15;           // A: M = row ; B: N = row
    const int half = lane >> 4;
    const int t0   = blockIdx.x * 16;

    // staging assignment: 2 lanes per row; lane covers 16 contiguous cols
    const int sr   = lane >> 1;           // staging row 0..15
    const int sc   = (lane & 1) * 16;     // staging col half: 0 or 16
    const float* srowp = S + (size_t)(t0 + sr) * NIN + sc;

    v8f c0 = {};   // output cols 0..15
    v8f c1 = {};   // output cols 16..31 (16..19 valid; B zero-padded)

    const int nFull = NIN / 32;           // full chunks (no bounds checks)
    for (int ch = 0; ch < nFull; ++ch) {
        const int kbase = ch * 32;
        // vectorized unguarded staging: 4x b128 load, cvt, 2x b128 ds store
        const v4f* p = (const v4f*)(srowp + kbase);
        const v4f f0 = p[0], f1 = p[1], f2 = p[2], f3 = p[3];
        v8h h0, h1;
        #pragma unroll
        for (int j = 0; j < 4; ++j) {
            h0[j]     = (_Float16)f0[j];
            h0[4 + j] = (_Float16)f1[j];
            h1[j]     = (_Float16)f2[j];
            h1[4 + j] = (_Float16)f3[j];
        }
        *(v8h*)&ldsA[sr * 32 + sc]     = h0;
        *(v8h*)&ldsA[sr * 32 + sc + 8] = h1;
        __syncthreads();

        wmma_step(ldsA, Bp + (size_t)ch * 1024, row, half, c0, c1);
        __syncthreads();
    }

    // tail chunk (partial K), guarded scalar staging, executed once
    if (nFull * 32 < NIN) {
        const int kbase = nFull * 32;
        const bool kvalid = (kbase + lane) < NIN;
        #pragma unroll
        for (int r2 = 0; r2 < 16; ++r2) {
            const float v = kvalid ? S[(size_t)(t0 + r2) * NIN + kbase + lane] : 0.f;
            ldsA[r2 * 32 + lane] = (_Float16)v;
        }
        __syncthreads();
        wmma_step(ldsA, Bp + (size_t)nFull * 1024, row, half, c0, c1);
    }

    // D layout: VGPR r, lane L -> M = r + 8*(L/16), N = L%16
    #pragma unroll
    for (int r = 0; r < 8; ++r) {
        const int m  = r + 8 * half;
        out[(size_t)(t0 + m) * kSUB + row] = c0[r];
        const int n1 = 16 + row;
        if (n1 < kSUB) out[(size_t)(t0 + m) * kSUB + n1] = c1[r];
    }
}

// =====================================================================
// Kernel 3: depthwise causal 200-tap convolution over time
// =====================================================================
__global__ __launch_bounds__(256) void conv_kernel(
    const float* __restrict__ in_e, const float* __restrict__ in_i,
    const float* __restrict__ kS,   const float* __restrict__ kNS,
    float* __restrict__ syn_s, float* __restrict__ syn_ns)
{
    const int idx = blockIdx.x * blockDim.x + threadIdx.x;
    if (idx >= kT_DATA * kSUB) return;
    const int t = idx / kSUB;
    const int s = idx % kSUB;
    const int mmax = (t < kT_SYN - 1) ? t : (kT_SYN - 1);
    float as = 0.f, ans = 0.f;
    for (int m = 0; m <= mmax; ++m) {
        const float xe = in_e[(size_t)(t - m) * kSUB + s];
        const float xi = in_i[(size_t)(t - m) * kSUB + s];
        as  += xe * kS [s * 400 + m] + xi * kS [s * 400 + 200 + m];
        ans += xe * kNS[s * 400 + m] + xi * kNS[s * 400 + 200 + m];
    }
    syn_s[idx]  = as;
    syn_ns[idx] = ans;
}

// =====================================================================
// Kernel 4: sequential scan, single wave32, lane s = subunit.
// Incremental pz buffer: pz shifts with zbuf => one 20-wide row/step.
// =====================================================================
__global__ __launch_bounds__(32) void scan_kernel(
    const float* __restrict__ syn_s, const float* __restrict__ syn_ns,
    const float* __restrict__ C_den,
    const float* __restrict__ hkSg, const float* __restrict__ hkNSg,
    const float* __restrict__ pkSg, const float* __restrict__ pkNSg,
    const float* __restrict__ W_s_sub, const float* __restrict__ W_ns_sub,
    const float* __restrict__ Theta_s, const float* __restrict__ Theta_ns,
    const float* __restrict__ V_o,
    float* __restrict__ V_out, float* __restrict__ Z_out)
{
    __shared__ float zbuf [kT_HIST * kSUB];
    __shared__ float pzbuf[kT_HIST * kSUB];
    __shared__ float hkS [kT_HIST * kSUB];
    __shared__ float hkNS[kT_HIST * kSUB];
    __shared__ float pkS [kT_HIST * kSUB];
    __shared__ float pkNS[kT_HIST * kSUB];
    __shared__ float ysh[kSUB], ynsh[kSUB], zsh[kSUB];

    const int lane = threadIdx.x;
    for (int i = lane; i < kT_HIST * kSUB; i += 32) {
        zbuf[i] = 0.f; pzbuf[i] = 0.f;
        hkS[i] = hkSg[i]; hkNS[i] = hkNSg[i];
        pkS[i] = pkSg[i]; pkNS[i] = pkNSg[i];
    }
    if (lane < kSUB) { ysh[lane] = 0.f; ynsh[lane] = 0.f; zsh[lane] = 0.f; }
    __syncthreads();

    float cden[kSUB];
    float wss = 0.f, wns = 0.f, ths = 0.f, thns = 0.f;
    if (lane < kSUB) {
        #pragma unroll
        for (int k = 0; k < kSUB; ++k) cden[k] = C_den[lane * kSUB + k];
        wss  = W_s_sub[lane];  wns  = W_ns_sub[lane];
        ths  = Theta_s[lane];  thns = Theta_ns[lane];
    }
    const float vo = V_o[0];

    int head = 0;   // physical index of logical tau=0 (oldest)
    for (int t = 0; t < kT_DATA; ++t) {
        float z = 0.f, ysn = 0.f, ynsn = 0.f;
        if (lane < kSUB) {
            float fsh = 0.f, fnsh = 0.f, fsp = 0.f, fnsp = 0.f;
            for (int tau = 0; tau < kT_HIST; ++tau) {
                int p = head + tau; if (p >= kT_HIST) p -= kT_HIST;
                const float zb  = zbuf [p * kSUB + lane];
                const float pzb = pzbuf[p * kSUB + lane];
                fsh  += zb  * hkS [tau * kSUB + lane];
                fnsh += zb  * hkNS[tau * kSUB + lane];
                fsp  += pzb * pkS [tau * kSUB + lane];
                fnsp += pzb * pkNS[tau * kSUB + lane];
            }
            float cys = 0.f, cyns = 0.f;
            #pragma unroll
            for (int k = 0; k < kSUB; ++k) {
                cys  += cden[k] * ysh[k];
                cyns += cden[k] * ynsh[k];
            }
            const float sd = syn_s [(size_t)t * kSUB + lane] + fsh  + fsp  + cys  - ths;
            const float nd = syn_ns[(size_t)t * kSUB + lane] + fnsh + fnsp + cyns - thns;
            z    = (sd >= 0.f) ? 1.f : 0.f;
            ysn  = z * wss;
            ynsn = tanhf(nd) * wns;
        }
        __syncthreads();
        if (lane < kSUB) zsh[lane] = z;
        __syncthreads();
        float pznew = 0.f;
        if (lane < kSUB) {
            #pragma unroll
            for (int k = 0; k < kSUB; ++k) pznew += cden[k] * zsh[k];
        }
        __syncthreads();
        if (lane < kSUB) {
            zbuf [head * kSUB + lane] = z;
            pzbuf[head * kSUB + lane] = pznew;
            ysh[lane]  = ysn;
            ynsh[lane] = ynsn;
            Z_out[(size_t)t * kSUB + lane] = z;
            if (lane == 0) V_out[t] = ynsn + vo;
        }
        __syncthreads();
        head = head + 1; if (head >= kT_HIST) head = 0;
    }
}

// =====================================================================
extern "C" void kernel_launch(void* const* d_in, const int* in_sizes, int n_in,
                              void* d_out, int out_size, void* d_ws, size_t ws_size,
                              hipStream_t stream) {
    const float* S_e       = (const float*)d_in[0];
    const float* S_i       = (const float*)d_in[1];
    const float* C_den     = (const float*)d_in[2];
    const float* C_syn_e   = (const float*)d_in[3];
    const float* C_syn_i   = (const float*)d_in[4];
    const float* W_s_syn   = (const float*)d_in[5];
    const float* W_ns_syn  = (const float*)d_in[6];
    const float* Tau_s_syn = (const float*)d_in[7];
    const float* Tau_ns_syn= (const float*)d_in[8];
    const float* D_s_syn   = (const float*)d_in[9];
    const float* D_ns_syn  = (const float*)d_in[10];
    const float* W_s_hist  = (const float*)d_in[11];
    const float* W_ns_hist = (const float*)d_in[12];
    const float* Tau_s_hist= (const float*)d_in[13];
    const float* Tau_ns_hist=(const float*)d_in[14];
    const float* W_s_prop  = (const float*)d_in[15];
    const float* W_ns_prop = (const float*)d_in[16];
    const float* Tau_s_prop= (const float*)d_in[17];
    const float* Tau_ns_prop=(const float*)d_in[18];
    const float* W_s_sub   = (const float*)d_in[19];
    const float* W_ns_sub  = (const float*)d_in[20];
    const float* Theta_s   = (const float*)d_in[21];
    const float* Theta_ns  = (const float*)d_in[22];
    const float* V_o       = (const float*)d_in[23];

    float* ws    = (float*)d_ws;
    float* in_e  = ws + WS_IN_E;
    float* in_i  = ws + WS_IN_I;
    float* synS  = ws + WS_SYN_S;
    float* synNS = ws + WS_SYN_NS;
    float* ksS   = ws + WS_KS;
    float* ksNS  = ws + WS_KNS;
    float* hkS   = ws + WS_HKS;
    float* hkNS  = ws + WS_HKNS;
    float* pkS   = ws + WS_PKS;
    float* pkNS  = ws + WS_PKNS;
    _Float16* BpE = (_Float16*)(ws + WS_BPE);
    _Float16* BpI = (_Float16*)(ws + WS_BPI);

    float* V_out = (float*)d_out;               // (10000,)
    float* Z_out = (float*)d_out + kT_DATA;     // (10000, 20)

    // 1) kernel tables
    genkern_kernel<<<32, 256, 0, stream>>>(
        W_s_syn, W_ns_syn, Tau_s_syn, Tau_ns_syn, D_s_syn, D_ns_syn,
        W_s_hist, W_ns_hist, Tau_s_hist, Tau_ns_hist,
        W_s_prop, W_ns_prop, Tau_s_prop, Tau_ns_prop,
        ksS, ksNS, hkS, hkNS, pkS, pkNS);

    // 2a) pack B operands into WMMA per-lane layout
    pack_b_kernel<<<(kChunksE * 1024 + 255) / 256, 256, 0, stream>>>(
        C_syn_e, BpE, kE_NO, kChunksE);
    pack_b_kernel<<<(kChunksI * 1024 + 255) / 256, 256, 0, stream>>>(
        C_syn_i, BpI, kI_NO, kChunksI);

    // 2b) WMMA one-hot projections (exact in f16)
    proj_wmma_kernel<<<kT_DATA / 16, 32, 0, stream>>>(S_e, BpE, in_e, kE_NO, kChunksE);
    proj_wmma_kernel<<<kT_DATA / 16, 32, 0, stream>>>(S_i, BpI, in_i, kI_NO, kChunksI);

    // 3) depthwise causal convolution (200 taps)
    conv_kernel<<<(kT_DATA * kSUB + 255) / 256, 256, 0, stream>>>(
        in_e, in_i, ksS, ksNS, synS, synNS);

    // 4) sequential scan, single wave
    scan_kernel<<<1, 32, 0, stream>>>(
        synS, synNS, C_den, hkS, hkNS, pkS, pkNS,
        W_s_sub, W_ns_sub, Theta_s, Theta_ns, V_o, V_out, Z_out);
}
